// MultiHeadCausalAttention_32796370272566
// MI455X (gfx1250) — matrix operand order repaired and checked by reading, not compiled, not run on hardware
//
#include <hip/hip_runtime.h>
#include <stdint.h>

// ---------------------------------------------------------------------------
// MI455X (gfx1250) multi-head causal attention, WMMA f16 (fp32 accumulate),
// CDNA5 async-to-LDS staging (ASYNCcnt), double-buffered GEMM pipeline,
// DPP-based softmax reductions, coalesced LDS-staged epilogues.
//
// Reference semantics honored, including both faithful bugs:
//   * scores are MULTIPLIED by sqrt(D)=32 (scores / D**-0.5) — folded into Q
//     fragments (power of two, exact in f16)
//   * attention output (B,H,T,hd) is viewed as (B,T,D) without transpose:
//     stored in (b,h,t,d) memory order, fed to the Wo GEMM as row-major.
// ---------------------------------------------------------------------------

typedef __attribute__((ext_vector_type(16))) _Float16 v16h;
typedef __attribute__((ext_vector_type(8)))  _Float16 v8h;
typedef __attribute__((ext_vector_type(4)))  _Float16 v4h;
typedef __attribute__((ext_vector_type(8)))  float    v8f;

#define DEV_INLINE __device__ __forceinline__

union F16Frag { v16h f; v8h h[2]; };

// ---- CDNA5 async copy: global -> LDS, 16B per lane, tracked by ASYNCcnt ----
DEV_INLINE void async_b128(const _Float16* g, _Float16* l) {
  const uint32_t lds_off = (uint32_t)(uintptr_t)l;  // low 32 bits = LDS addr
  asm volatile("global_load_async_to_lds_b128 %0, %1, off"
               :: "v"(lds_off), "v"(g) : "memory");
}
DEV_INLINE void wait_async0() {
  asm volatile("s_wait_asynccnt 0" ::: "memory");
}
DEV_INLINE void wait_ds0() {
  asm volatile("s_wait_dscnt 0" ::: "memory");
}

// ---- DPP-based 16-lane reductions (xor1/2/4 via DPP8, xor8 via row_ror:8) --
constexpr int dpp8sel(int a, int b, int c, int d, int e, int f, int g, int h) {
  return a | (b << 3) | (c << 6) | (d << 9) | (e << 12) | (f << 15) |
         (g << 18) | (h << 21);
}
template <int SEL>
DEV_INLINE float dpp8_f(float x) {
  return __int_as_float(__builtin_amdgcn_mov_dpp8(__float_as_int(x), SEL));
}
DEV_INLINE float dpp_ror8row(float x) {  // lane i <- lane i^8 within 16-row
  return __int_as_float(
      __builtin_amdgcn_update_dpp(0, __float_as_int(x), 0x128, 0xf, 0xf, true));
}
DEV_INLINE float rmax16(float x) {
  x = fmaxf(x, dpp8_f<dpp8sel(1, 0, 3, 2, 5, 4, 7, 6)>(x));
  x = fmaxf(x, dpp8_f<dpp8sel(2, 3, 0, 1, 6, 7, 4, 5)>(x));
  x = fmaxf(x, dpp8_f<dpp8sel(4, 5, 6, 7, 0, 1, 2, 3)>(x));
  x = fmaxf(x, dpp_ror8row(x));
  return x;
}
DEV_INLINE float rsum16(float x) {
  x += dpp8_f<dpp8sel(1, 0, 3, 2, 5, 4, 7, 6)>(x);
  x += dpp8_f<dpp8sel(2, 3, 0, 1, 6, 7, 4, 5)>(x);
  x += dpp8_f<dpp8sel(4, 5, 6, 7, 0, 1, 2, 3)>(x);
  x += dpp_ror8row(x);
  return x;
}

// A-fragment (16x32 f16, M x K) per CDNA5 ISA layout:
// lane L: row = L&15, half hl = L>>4; elems 0..7 -> k = hl*8+e, 8..15 -> 16+hl*8+(e-8)
DEV_INLINE v16h load_a_frag(const _Float16* tile, int ld, int lane) {
  const int r = lane & 15, hl = lane >> 4;
  F16Frag u;
  u.h[0] = *(const v8h*)(tile + (size_t)r * ld + hl * 8);
  u.h[1] = *(const v8h*)(tile + (size_t)r * ld + 16 + hl * 8);
  return u.f;
}

// B-fragment (32x16 f16, K x N) from tile stored [n][k] row-major:
// lane L: col n = L&15, half hl = L>>4; elems 0..7 -> k = hl*8+e, 8..15 -> 16+hl*8+(e-8)
DEV_INLINE v16h load_b_frag(const _Float16* tile, int ld, int lane) {
  const int n = lane & 15, hl = lane >> 4;
  F16Frag u;
  u.h[0] = *(const v8h*)(tile + (size_t)n * ld + hl * 8);
  u.h[1] = *(const v8h*)(tile + (size_t)n * ld + 16 + hl * 8);
  return u.f;
}

DEV_INLINE v8f wmma_f16(v16h a, v16h b, v8f c) {
  return __builtin_amdgcn_wmma_f32_16x16x32_f16(false, a, false, b, (short)0, c,
                                                false, false);
}

// ---------------------------------------------------------------------------
// f32 -> f16 conversion, 4 elems/thread (b128 load, b64 store)
// ---------------------------------------------------------------------------
__global__ void cvt_f32_f16(const float* __restrict__ in,
                            _Float16* __restrict__ out, int n) {
  int i = (blockIdx.x * blockDim.x + threadIdx.x) * 4;
  if (i < n) {
    float4 v = *(const float4*)(in + i);
    v4h r = { (_Float16)v.x, (_Float16)v.y, (_Float16)v.z, (_Float16)v.w };
    *(v4h*)(out + i) = r;
  }
}

// ---------------------------------------------------------------------------
// C[m,n] = sum_k A[m,k] * W[n,k]   (i.e. A @ W.T, both row-major f16)
// 256 threads = 8 waves, 128x128 workgroup tile, K-step 64.
// Double-buffered LDS staged with global_load_async_to_lds_b128 (next step's
// copies overlap the current step's 16 WMMAs). One 64KB LDS block is reused
// as the C staging tile for a fully coalesced b128 epilogue.
// Wave grid 2x4: wave (wm,wn) owns a 64x32 region = 4x2 WMMA tiles.
// ---------------------------------------------------------------------------
template <typename OutT>
__global__ __launch_bounds__(256) void gemm_nt(const _Float16* __restrict__ A,
                                               const _Float16* __restrict__ W,
                                               OutT* __restrict__ C,
                                               int M, int N, int K) {
  __shared__ __align__(16) unsigned char smem[65536];
  _Float16* At = (_Float16*)smem;            // [2][128*64]  32 KB
  _Float16* Wt = (_Float16*)(smem + 32768);  // [2][128*64]  32 KB
  OutT* Ct = (OutT*)smem;                    // reused after the K loop

  const int tid  = threadIdx.x;
  const int lane = tid & 31;
  const int wave = tid >> 5;
  const int wm   = wave >> 2;      // 0..1  (row group of 64)
  const int wn   = wave & 3;       // 0..3  (col group of 32)
  const int bm   = blockIdx.x * 128;
  const int bn   = blockIdx.y * 128;

  v8f acc[4][2] = {};
  int phase = 0;

  // Prologue: stage K-step 0 into buffer 0 (128 rows x 64 halfs per tile).
  #pragma unroll
  for (int i = 0; i < 4; ++i) {
    const int ch = tid + i * 256;          // 0..1023
    const int r  = ch >> 3;
    const int c8 = (ch & 7) * 8;
    async_b128(A + (size_t)(bm + r) * K + c8, At + r * 64 + c8);
    async_b128(W + (size_t)(bn + r) * K + c8, Wt + r * 64 + c8);
  }
  wait_async0();
  __syncthreads();

  for (int kb = 0; kb < K; kb += 64) {
    // Prefetch next K-step into the other buffer (overlaps the WMMAs below).
    if (kb + 64 < K) {
      const int nb = (phase ^ 1) * (128 * 64);
      #pragma unroll
      for (int i = 0; i < 4; ++i) {
        const int ch = tid + i * 256;
        const int r  = ch >> 3;
        const int c8 = (ch & 7) * 8;
        async_b128(A + (size_t)(bm + r) * K + kb + 64 + c8, At + nb + r * 64 + c8);
        async_b128(W + (size_t)(bn + r) * K + kb + 64 + c8, Wt + nb + r * 64 + c8);
      }
    }

    const _Float16* Ab = At + phase * (128 * 64);
    const _Float16* Wb = Wt + phase * (128 * 64);
    #pragma unroll
    for (int c = 0; c < 2; ++c) {          // two K=32 halves of the 64-step
      v16h af[4], bf[2];
      #pragma unroll
      for (int i = 0; i < 4; ++i)
        af[i] = load_a_frag(Ab + (wm * 64 + i * 16) * 64 + c * 32, 64, lane);
      #pragma unroll
      for (int j = 0; j < 2; ++j)
        bf[j] = load_b_frag(Wb + (wn * 32 + j * 16) * 64 + c * 32, 64, lane);
      #pragma unroll
      for (int i = 0; i < 4; ++i)
        #pragma unroll
        for (int j = 0; j < 2; ++j)
          acc[i][j] = wmma_f16(af[i], bf[j], acc[i][j]);
    }

    wait_async0();
    __syncthreads();
    phase ^= 1;
  }

  // Epilogue: restage accumulators in LDS (C layout), then coalesced stores.
  // C/D layout: lane L, vgpr v -> row = v + 8*(L>>4), col = L&15
  const int hl = lane >> 4, nl = lane & 15;
  #pragma unroll
  for (int i = 0; i < 4; ++i)
    #pragma unroll
    for (int j = 0; j < 2; ++j)
      #pragma unroll
      for (int v = 0; v < 8; ++v)
        Ct[(wm * 64 + i * 16 + v + 8 * hl) * 128 + wn * 32 + j * 16 + nl] =
            (OutT)acc[i][j][v];
  __syncthreads();

  if constexpr (sizeof(OutT) == 2) {
    const _Float16* Cs = (const _Float16*)Ct;
    #pragma unroll
    for (int i = 0; i < 8; ++i) {
      const int ch = tid + i * 256;        // 2048 chunks of 8 halfs
      const int r  = ch >> 4;
      const int c8 = (ch & 15) * 8;
      *(v8h*)((_Float16*)C + (size_t)(bm + r) * N + bn + c8) =
          *(const v8h*)(Cs + r * 128 + c8);
    }
  } else {
    const float* Cs = (const float*)Ct;
    #pragma unroll
    for (int i = 0; i < 16; ++i) {
      const int ch = tid + i * 256;        // 4096 chunks of 4 floats
      const int r  = ch >> 5;
      const int c4 = (ch & 31) * 4;
      *(float4*)((float*)C + (size_t)(bm + r) * N + bn + c4) =
          *(const float4*)(Cs + r * 128 + c4);
    }
  }
}

// ---------------------------------------------------------------------------
// Flash-style causal attention. Grid: (T/64, H, B); 128 threads = 4 waves.
// Each wave owns a 16-row query tile; 32-key steps staged in LDS (K tile via
// async-to-LDS, V tile transposed through VGPRs).
// Q/K/V layout: (B, T, H, HD) row-major = GEMM output (B*T, D).
// Output written in (b, h, t, d) order (the reference's reshape bug).
// ---------------------------------------------------------------------------
__global__ __launch_bounds__(128) void attn_fused(const _Float16* __restrict__ Q,
                                                  const _Float16* __restrict__ Kg,
                                                  const _Float16* __restrict__ Vg,
                                                  _Float16* __restrict__ O) {
  constexpr int T = 2048, D = 1024, HD = 64, NH = 16;

  __shared__ __align__(16) _Float16 Kl[32 * 64];      // [key][d]
  __shared__ __align__(16) _Float16 Vl[64 * 32];      // [d][key] (transposed)
  __shared__ __align__(16) _Float16 Pl[4][16 * 32];   // per-wave p tile
  __shared__ __align__(16) _Float16 Ol[4][16 * 64];   // per-wave out staging

  const int tid  = threadIdx.x;
  const int lane = tid & 31;
  const int wave = tid >> 5;
  const int hl   = lane >> 4;
  const int nl   = lane & 15;
  const int b    = blockIdx.z;
  const int h    = blockIdx.y;
  const int qb   = blockIdx.x * 64;
  const int qrow0 = qb + wave * 16;
  const int qlast = qrow0 + 15;

  // Preload this wave's Q A-fragments (16 rows x 64 d = two K=32 fragments)
  // and fold the faithful-bug scale sqrt(D)=32 into them (exact: power of 2).
  const _Float16* qptr = Q + ((size_t)b * T + qrow0) * D + h * HD;
  F16Frag q0, q1;
  q0.h[0] = *(const v8h*)(qptr + (size_t)nl * D + hl * 8);
  q0.h[1] = *(const v8h*)(qptr + (size_t)nl * D + 16 + hl * 8);
  q1.h[0] = *(const v8h*)(qptr + (size_t)nl * D + 32 + hl * 8);
  q1.h[1] = *(const v8h*)(qptr + (size_t)nl * D + 48 + hl * 8);
  #pragma unroll
  for (int e = 0; e < 16; ++e) {
    q0.f[e] = q0.f[e] * (_Float16)32.0f;
    q1.f[e] = q1.f[e] * (_Float16)32.0f;
  }

  float mrow[8], lrow[8];
  #pragma unroll
  for (int v = 0; v < 8; ++v) { mrow[v] = -__builtin_inff(); lrow[v] = 0.0f; }
  v8f accv[4] = {};

  for (int kb = 0; kb < qb + 64; kb += 32) {
    // ---- cooperative staging (uniform across all 4 waves) ----
    // K block: 32 rows x 64 halfs, row-major; async global->LDS b128.
    #pragma unroll
    for (int t = tid; t < 256; t += 128) {
      const int row = t >> 3;
      const int ch  = (t & 7) * 8;
      async_b128(Kg + ((size_t)b * T + kb + row) * D + h * HD + ch,
                 Kl + row * 64 + ch);
    }
    // V block transposed into [d][key] (VGPR-mediated transpose).
    #pragma unroll
    for (int t = tid; t < 256; t += 128) {
      const int kl = t & 31;
      const int dg = t >> 5;
      v8h tmp = *(const v8h*)(Vg + ((size_t)b * T + kb + kl) * D + h * HD + dg * 8);
      #pragma unroll
      for (int j = 0; j < 8; ++j) Vl[(dg * 8 + j) * 32 + kl] = tmp[j];
    }
    wait_async0();
    __syncthreads();

    if (kb <= qlast) {  // wave-uniform: skip key blocks fully in the future
      // ---- scores: 16 q-rows x 32 keys, contraction over d=64 ----
      v8f s0 = {}, s1 = {};
      s0 = wmma_f16(q0.f, load_b_frag(Kl, 64, lane), s0);
      s0 = wmma_f16(q1.f, load_b_frag(Kl + 32, 64, lane), s0);
      s1 = wmma_f16(q0.f, load_b_frag(Kl + 16 * 64, 64, lane), s1);
      s1 = wmma_f16(q1.f, load_b_frag(Kl + 16 * 64 + 32, 64, lane), s1);

      // Masking only needed for blocks overlapping the diagonal.
      const bool needMask = (kb + 31 > qrow0);

      // ---- online softmax, row r = v + 8*hl lives in one 16-lane half ----
      _Float16* pw = Pl[wave];
      #pragma unroll
      for (int v = 0; v < 8; ++v) {
        const int rq = qrow0 + v + 8 * hl;
        float x0 = s0[v];
        float x1 = s1[v];
        if (needMask) {
          if (kb + nl > rq)      x0 = -__builtin_inff();
          if (kb + 16 + nl > rq) x1 = -__builtin_inff();
        }
        const float mx = rmax16(fmaxf(x0, x1));
        const float nm = fmaxf(mrow[v], mx);
        const float alpha = __expf(mrow[v] - nm);
        const float p0 = __expf(x0 - nm);
        const float p1 = __expf(x1 - nm);
        lrow[v] = lrow[v] * alpha + rsum16(p0 + p1);
        mrow[v] = nm;
        #pragma unroll
        for (int c = 0; c < 4; ++c) accv[c][v] = accv[c][v] * alpha;
        pw[(v + 8 * hl) * 32 + nl]      = (_Float16)p0;
        pw[(v + 8 * hl) * 32 + 16 + nl] = (_Float16)p1;
      }

      // C-layout -> A-layout relayout via LDS round trip (same-wave region).
      wait_ds0();
      const v16h pf = load_a_frag(pw, 32, lane);

      // ---- accv += p (16x32) @ V (32 x 64), 4 d-chunks of 16 ----
      #pragma unroll
      for (int c = 0; c < 4; ++c)
        accv[c] = wmma_f16(pf, load_b_frag(Vl + (c * 16) * 32, 32, lane), accv[c]);
    }
    __syncthreads();
  }

  // ---- normalize, restage per-wave in LDS, coalesced b128 stores ----
  float inv[8];
  #pragma unroll
  for (int v = 0; v < 8; ++v) inv[v] = 1.0f / lrow[v];
  _Float16* ow = Ol[wave];
  #pragma unroll
  for (int c = 0; c < 4; ++c)
    #pragma unroll
    for (int v = 0; v < 8; ++v)
      ow[(v + 8 * hl) * 64 + c * 16 + nl] = (_Float16)(accv[c][v] * inv[v]);
  wait_ds0();

  // 16 rows x 64 halfs per wave -> 128 b128 chunks, 4 per lane.
  _Float16* obase = O + (((size_t)b * NH + h) * T + qrow0) * HD;
  #pragma unroll
  for (int i = 0; i < 4; ++i) {
    const int cc = lane + 32 * i;          // 0..127
    const int r  = cc >> 3;
    const int d8 = (cc & 7) * 8;
    *(v8h*)(obase + (size_t)r * HD + d8) = *(const v8h*)(ow + r * 64 + d8);
  }
}

// ---------------------------------------------------------------------------
// Host-side launch
// ---------------------------------------------------------------------------
extern "C" void kernel_launch(void* const* d_in, const int* in_sizes, int n_in,
                              void* d_out, int out_size, void* d_ws, size_t ws_size,
                              hipStream_t stream) {
  (void)in_sizes; (void)n_in; (void)out_size; (void)ws_size;

  const float* x  = (const float*)d_in[0];
  const float* Wq = (const float*)d_in[1];
  const float* Wk = (const float*)d_in[2];
  const float* Wv = (const float*)d_in[3];
  const float* Wo = (const float*)d_in[4];
  float* out = (float*)d_out;

  const int NX = 4 * 2048 * 1024;   // 8388608 elems of x
  const int NW = 1024 * 1024;       // per weight

  char* ws = (char*)d_ws;
  _Float16* xh  = (_Float16*)(ws);                        // 16 MB
  _Float16* wqh = (_Float16*)(ws + (size_t)16 * 1048576); //  2 MB each
  _Float16* wkh = (_Float16*)(ws + (size_t)18 * 1048576);
  _Float16* wvh = (_Float16*)(ws + (size_t)20 * 1048576);
  _Float16* woh = (_Float16*)(ws + (size_t)22 * 1048576);
  _Float16* Qb  = (_Float16*)(ws + (size_t)24 * 1048576); // 16 MB each
  _Float16* Kb  = (_Float16*)(ws + (size_t)40 * 1048576);
  _Float16* Vb  = (_Float16*)(ws + (size_t)56 * 1048576);
  _Float16* Ab  = (_Float16*)(ws + (size_t)72 * 1048576); // ends at 88 MB

  // 1) fp32 -> fp16 once (halves all GEMM read traffic afterwards)
  cvt_f32_f16<<<dim3(NX / 1024), dim3(256), 0, stream>>>(x,  xh,  NX);
  cvt_f32_f16<<<dim3(NW / 1024), dim3(256), 0, stream>>>(Wq, wqh, NW);
  cvt_f32_f16<<<dim3(NW / 1024), dim3(256), 0, stream>>>(Wk, wkh, NW);
  cvt_f32_f16<<<dim3(NW / 1024), dim3(256), 0, stream>>>(Wv, wvh, NW);
  cvt_f32_f16<<<dim3(NW / 1024), dim3(256), 0, stream>>>(Wo, woh, NW);

  // 2) Q/K/V projections: (8192x1024) @ (1024x1024)^T, WMMA f16
  gemm_nt<_Float16><<<dim3(64, 8), dim3(256), 0, stream>>>(xh, wqh, Qb, 8192, 1024, 1024);
  gemm_nt<_Float16><<<dim3(64, 8), dim3(256), 0, stream>>>(xh, wkh, Kb, 8192, 1024, 1024);
  gemm_nt<_Float16><<<dim3(64, 8), dim3(256), 0, stream>>>(xh, wvh, Vb, 8192, 1024, 1024);

  // 3) fused causal attention (flash-style, WMMA for both batched GEMMs)
  attn_fused<<<dim3(32, 16, 4), dim3(128), 0, stream>>>(Qb, Kb, Vb, Ab);

  // 4) output projection to fp32 d_out (Ab already in the buggy (b,h,t,d) view)
  gemm_nt<float><<<dim3(64, 8), dim3(256), 0, stream>>>(Ab, woh, out, 8192, 1024, 1024);
}
